// SwinAttnBlock_76785425318149
// MI455X (gfx1250) — matrix operand order repaired and checked by reading, not compiled, not run on hardware
//
#include <hip/hip_runtime.h>
#include <cstdint>

typedef __attribute__((ext_vector_type(16))) _Float16 v16h;
typedef __attribute__((ext_vector_type(8)))  float    v8f;

#define HGT   448
#define WID   448
#define CH    192
#define HD    32
#define NTOK  49
#define NPAD  64

// LDS strides (element units)
#define SX_STRIDE  200   // halves; 400B/row -> 36 banks/row, conflict-free for 16 rows
#define SQK_STRIDE 392   // halves; Q cols 0..191, K cols 192..383
#define SVT_STRIDE 72    // halves; V transposed: [channel][token]
#define SS_STRIDE  65    // floats; raw scores (3 heads/batch)
#define SP_STRIDE  66    // halves; softmax probs (3 heads/batch)
#define SW_STRIDE  200   // halves; per-wave weight tile stage

#define SX_ELEMS   (64 * SX_STRIDE)      // 12800 h
#define SQK_ELEMS  (64 * SQK_STRIDE)     // 25088 h
#define SVT_ELEMS  (192 * SVT_STRIDE)    // 13824 h
#define SS_ELEMS   (3 * 64 * SS_STRIDE)  // 12480 f
#define SP_ELEMS   (3 * 64 * SP_STRIDE)  // 12672 h
#define SW_ELEMS   (8 * 16 * SW_STRIDE)  // 25600 h

#define SMEM_BYTES (((SX_ELEMS + SQK_ELEMS + SVT_ELEMS + SP_ELEMS + SW_ELEMS) * 2) + SS_ELEMS * 4)

static_assert(SMEM_BYTES <= 320 * 1024, "LDS budget exceeds 320KB WGP limit");

__device__ __forceinline__ v8f wmma16(v16h a, v16h b, v8f c) {
  // D = A(16x32 f16) x B(32x16 f16) + C(16x16 f32)
  return __builtin_amdgcn_wmma_f32_16x16x32_f16(false, a, false, b, (short)0, c,
                                                false, false);
}

// A-fragment from row-major [m][k] f16 LDS. ISA 16-bit A 16x32 layout:
// lane(lo)=row; k = (i&3)*2 + hi*8 + (i>=4 ? 16 : 0), pairs packed per dword.
__device__ __forceinline__ v16h fragA(const _Float16* base, int stride,
                                      int row0, int col0, int lo, int hi) {
  const _Float16* p = base + (row0 + lo) * stride + col0;
  union { unsigned u[8]; v16h v; } r;
#pragma unroll
  for (int i = 0; i < 8; ++i) {
    int k = ((i & 4) ? 16 : 0) + hi * 8 + (i & 3) * 2;
    r.u[i] = *(const unsigned*)(p + k);
  }
  return r.v;
}

// B-fragment of B(32x16) taken from [n][k] row-major (i.e. B^T row-major) LDS.
// ISA B layout: lane(lo)=column n; vgpr i holds k = (i>=4 ? 16 : 0) + hi*8 + (i&3)*2 (+pair).
__device__ __forceinline__ v16h fragB(const _Float16* base, int stride,
                                      int n0, int k0, int lo, int hi) {
  const _Float16* p = base + (n0 + lo) * stride + k0;
  union { unsigned u[8]; v16h v; } r;
#pragma unroll
  for (int i = 0; i < 8; ++i) {
    int k = ((i & 4) ? 16 : 0) + hi * 8 + (i & 3) * 2;
    r.u[i] = *(const unsigned*)(p + k);
  }
  return r.v;
}

// Region id for shifted-window mask. H=W=448, WS=7, SS=3:
// rows [0,441)->0, [441,445)->1, [445,448)->2 (same for cols).
// Only the last window row/col (wh==63 / ww==63) straddles regions.
__device__ __forceinline__ int region_id(int tok, int wh, int ww) {
  int i = tok / 7;
  int j = tok - i * 7;
  int rr = (wh == 63) ? ((i < 4) ? 1 : 2) : 0;
  int rc = (ww == 63) ? ((j < 4) ? 1 : 2) : 0;
  return rr * 3 + rc;
}

__global__ __launch_bounds__(256) void swin_attn_block(
    const float* __restrict__ x, const float* __restrict__ qkv_w,
    const float* __restrict__ qkv_b, const float* __restrict__ proj_w,
    const float* __restrict__ proj_b, float* __restrict__ out) {
  extern __shared__ char smem_raw[];
  _Float16* sX  = (_Float16*)smem_raw;     // window activations (later reused as sO)
  _Float16* sQK = sX + SX_ELEMS;           // Q (cols 0..191, pre-scaled), K (192..383)
  _Float16* sVT = sQK + SQK_ELEMS;         // V transposed [channel][token]
  float*    sS  = (float*)(sVT + SVT_ELEMS);
  _Float16* sP  = (_Float16*)(sS + SS_ELEMS);
  _Float16* sWt = sP + SP_ELEMS;           // per-wave weight tiles
  _Float16* sO  = sX;                      // attention output, pre-projection

  const int tid  = threadIdx.x;
  const int wave = tid >> 5;
  const int lane = tid & 31;
  const int lo = lane & 15;
  const int hi = lane >> 4;

  const int widx = blockIdx.x;      // 0..8191
  const int bb = widx >> 12;        // batch
  const int wi = widx & 4095;
  const int wh = wi >> 6;
  const int ww = wi & 63;

  const v8f vzero = {0.f, 0.f, 0.f, 0.f, 0.f, 0.f, 0.f, 0.f};

  // ---- Phase 0: gather shifted window -> sX (f16), zero-pad rows 49..63 ----
  for (int t = tid; t < NPAD * CH; t += 256) {
    int r = t / CH, c = t - r * CH;
    _Float16 v = (_Float16)0.f;
    if (r < NTOK) {
      int i = r / 7, j = r - i * 7;
      int hs = wh * 7 + i + 3; if (hs >= HGT) hs -= HGT;  // roll(-3) gather
      int ws = ww * 7 + j + 3; if (ws >= WID) ws -= WID;
      v = (_Float16)x[(bb * (HGT * WID) + hs * WID + ws) * CH + c];
    }
    sX[r * SX_STRIDE + c] = v;
  }
  __syncthreads();

  // ---- Phase 1: QKV = X(64x192) @ qkv_w^T(192x576) + b; Q pre-scaled ----
  _Float16* wbuf = sWt + wave * 16 * SW_STRIDE;
  for (int ct = wave; ct < 36; ct += 8) {
    // stage 16 weight rows (o = ct*16..+15) as f16 [o][c] (== B^T row-major)
    for (int t = lane; t < 16 * CH; t += 32) {
      int o = t / CH, c = t - o * CH;
      wbuf[o * SW_STRIDE + c] = (_Float16)qkv_w[(ct * 16 + o) * CH + c];
    }
    if (ct + 8 < 36)  // warm L2->L0 for next tile (global_prefetch_b8)
      __builtin_prefetch(&qkv_w[((ct + 8) * 16 + lo) * CH + hi * 96], 0, 0);

    v8f acc[4] = {vzero, vzero, vzero, vzero};
#pragma unroll
    for (int ks = 0; ks < 6; ++ks) {
      v16h bf = fragB(wbuf, SW_STRIDE, 0, ks * 32, lo, hi);
#pragma unroll
      for (int mt = 0; mt < 4; ++mt) {
        v16h af = fragA(sX, SX_STRIDE, mt * 16, ks * 32, lo, hi);
        acc[mt] = wmma16(af, bf, acc[mt]);
      }
    }
    float bias = qkv_b[ct * 16 + lo];
    const float scl = (ct < 12) ? 0.17677669529663687f : 1.0f;  // 1/sqrt(32) on Q
#pragma unroll
    for (int mt = 0; mt < 4; ++mt) {
#pragma unroll
      for (int r = 0; r < 8; ++r) {
        int row = mt * 16 + r + hi * 8;
        int col = ct * 16 + lo;
        float v = (acc[mt][r] + bias) * scl;
        if (ct < 24) sQK[row * SQK_STRIDE + col] = (_Float16)v;
        else         sVT[(col - 384) * SVT_STRIDE + row] = (_Float16)v;  // V^T
      }
    }
  }
  __syncthreads();

  // ---- Phase 2: attention, 2 batches of 3 heads ----
  const bool edge = (wh == 63) || (ww == 63);
  for (int hb = 0; hb < 2; ++hb) {
    int h0 = hb * 3;
    // S = Q @ K^T  (K dim = head_dim = 32 -> single WMMA per 16x16 tile)
    for (int t = wave; t < 48; t += 8) {
      int hl = t >> 4, mt = (t >> 2) & 3, nt = t & 3;
      int h = h0 + hl;
      v16h af = fragA(sQK, SQK_STRIDE, mt * 16, h * HD, lo, hi);
      v16h bf = fragB(sQK, SQK_STRIDE, nt * 16, CH + h * HD, lo, hi);
      v8f sc = wmma16(af, bf, vzero);
      int colr = nt * 16 + lo;
      int ridc = (edge && colr < NTOK) ? region_id(colr, wh, ww) : 0;
#pragma unroll
      for (int r = 0; r < 8; ++r) {
        int row = mt * 16 + r + hi * 8;
        float v = sc[r];
        if (colr >= NTOK) v = -1e30f;                        // kill padded keys
        else if (edge && row < NTOK && region_id(row, wh, ww) != ridc)
          v -= 100.f;                                        // shift mask
        sS[(hl * 64 + row) * SS_STRIDE + colr] = v;
      }
    }
    __syncthreads();
    // row softmax (192 rows)
    if (tid < 192) {
      float* srow = sS + tid * SS_STRIDE;
      float mx = -1e30f;
      for (int m = 0; m < 64; ++m) mx = fmaxf(mx, srow[m]);
      float sum = 0.f;
      for (int m = 0; m < 64; ++m) sum += __expf(srow[m] - mx);
      float inv = 1.f / sum;
      _Float16* prow = sP + tid * SP_STRIDE;
      for (int m = 0; m < 64; ++m)
        prow[m] = (_Float16)(__expf(srow[m] - mx) * inv);
    }
    __syncthreads();
    // O = P @ V  (A = probs row-major, B = V^T row-major == V column-major)
    for (int t = wave; t < 24; t += 8) {
      int hl = t >> 3, mt = (t >> 1) & 3, c2 = t & 1;
      v8f acc = vzero;
#pragma unroll
      for (int ks = 0; ks < 2; ++ks) {
        v16h af = fragA(sP, SP_STRIDE, hl * 64 + mt * 16, ks * 32, lo, hi);
        v16h bf = fragB(sVT, SVT_STRIDE, (h0 + hl) * HD + c2 * 16, ks * 32, lo, hi);
        acc = wmma16(af, bf, acc);
      }
#pragma unroll
      for (int r = 0; r < 8; ++r) {
        int row = mt * 16 + r + hi * 8;
        int col = (h0 + hl) * HD + c2 * 16 + lo;
        sO[row * SX_STRIDE + col] = (_Float16)acc[r];
      }
    }
    __syncthreads();
  }

  // ---- Phase 3: proj + bias + residual, scatter with reverse roll ----
  for (int ct = wave; ct < 12; ct += 8) {
    for (int t = lane; t < 16 * CH; t += 32) {
      int o = t / CH, c = t - o * CH;
      wbuf[o * SW_STRIDE + c] = (_Float16)proj_w[(ct * 16 + o) * CH + c];
    }
    v8f acc[4] = {vzero, vzero, vzero, vzero};
#pragma unroll
    for (int ks = 0; ks < 6; ++ks) {
      v16h bf = fragB(wbuf, SW_STRIDE, 0, ks * 32, lo, hi);
#pragma unroll
      for (int mt = 0; mt < 4; ++mt) {
        v16h af = fragA(sO, SX_STRIDE, mt * 16, ks * 32, lo, hi);
        acc[mt] = wmma16(af, bf, acc[mt]);
      }
    }
    float bias = proj_b[ct * 16 + lo];
#pragma unroll
    for (int mt = 0; mt < 4; ++mt) {
#pragma unroll
      for (int r = 0; r < 8; ++r) {
        int row = mt * 16 + r + hi * 8;
        if (row >= NTOK) continue;
        int i = row / 7, j = row - i * 7;
        int oh = wh * 7 + i + 3; if (oh >= HGT) oh -= HGT;  // roll(+3) scatter
        int ow = ww * 7 + j + 3; if (ow >= WID) ow -= WID;
        int col = ct * 16 + lo;
        int addr = (bb * (HGT * WID) + oh * WID + ow) * CH + col;
        out[addr] = x[addr] + acc[mt][r] + bias;            // residual at same index
      }
    }
  }
}

extern "C" void kernel_launch(void* const* d_in, const int* in_sizes, int n_in,
                              void* d_out, int out_size, void* d_ws, size_t ws_size,
                              hipStream_t stream) {
  (void)in_sizes; (void)n_in; (void)out_size; (void)d_ws; (void)ws_size;
  const float* x      = (const float*)d_in[0];
  const float* qkv_w  = (const float*)d_in[1];
  const float* qkv_b  = (const float*)d_in[2];
  const float* proj_w = (const float*)d_in[3];
  const float* proj_b = (const float*)d_in[4];
  float* out = (float*)d_out;

  (void)hipFuncSetAttribute((const void*)swin_attn_block,
                            hipFuncAttributeMaxDynamicSharedMemorySize,
                            (int)SMEM_BYTES);
  dim3 grid(2 * 64 * 64);  // 8192 windows, one workgroup each
  dim3 block(256);         // 8 wave32
  swin_attn_block<<<grid, block, SMEM_BYTES, stream>>>(x, qkv_w, qkv_b, proj_w,
                                                       proj_b, out);
}
